// TransformerBlock_20435454394582
// MI455X (gfx1250) — matrix operand order repaired and checked by reading, not compiled, not run on hardware
//
#include <hip/hip_runtime.h>
#include <math.h>

// ---------------------------------------------------------------------------
// MI455X / gfx1250 transformer block.
// B=8, S=1024, H=1024, NH=16, HD=64, MLP=4096.  f32 in/out, bf16 WMMA inside.
// GEMM tiles are fed by the Tensor Data Mover (tensor_load_to_lds, TENSORcnt),
// double-buffered in LDS; attention is a WMMA flash kernel.
// ---------------------------------------------------------------------------

#define BB   8
#define SS   1024
#define HH   1024
#define NHH  16
#define HDD  64
#define MLPD 4096
#define MM   (BB*SS)      // 8192 rows

typedef __attribute__((ext_vector_type(16))) __bf16 bf16x16;
typedef __attribute__((ext_vector_type(8)))  __bf16 bf16x8;
typedef __attribute__((ext_vector_type(4)))  __bf16 bf16x4;
typedef __attribute__((ext_vector_type(8)))  float  f32x8;
typedef __attribute__((ext_vector_type(4)))  unsigned int uint32x4;
typedef __attribute__((ext_vector_type(4)))  int    int32x4;
typedef __attribute__((ext_vector_type(8)))  int    int32x8;

__device__ __forceinline__ bf16x16 cat8(bf16x8 lo, bf16x8 hi) {
  return __builtin_shufflevector(lo, hi, 0,1,2,3,4,5,6,7,8,9,10,11,12,13,14,15);
}
__device__ __forceinline__ f32x8 wmma_bf16(bf16x16 a, bf16x16 b, f32x8 c) {
  // D = A(16x32 bf16) * B(32x16 bf16) + C(16x16 f32)
  return __builtin_amdgcn_wmma_f32_16x16x32_bf16(false, a, false, b,
                                                 (short)0, c, false, false);
}

// ---------------------------------------------------------------------------
// TDM: DMA one [128 rows x 32 cols] bf16 tile (row stride = strideElems) from
// global into LDS at byte offset ldsAddr, rows packed contiguously (row-major
// [128][32]).  2D descriptor; tensor scoped to the tile so no OOB can occur.
// One issue per executing wave; completion tracked with TENSORcnt.
// ---------------------------------------------------------------------------
__device__ __forceinline__ void tdm_load_tile_128x32(const void* gsrc,
                                                     unsigned ldsAddr,
                                                     int strideElems) {
  const unsigned long long ga = (unsigned long long)(uintptr_t)gsrc;
  // Group 0: count=1 | lds_addr | global_addr[56:0] | type=2 ("image")
  uint32x4 g0 = { 1u,
                  ldsAddr,
                  (unsigned)(ga & 0xffffffffu),
                  (unsigned)((ga >> 32) & 0x01ffffffu) | (2u << 30) };
  // Group 1: data_size=2B; tensor_dim0=32, tensor_dim1=128;
  //          tile_dim0=32, tile_dim1=128; tensor_dim0_stride=strideElems.
  int32x8 g1 = { 0x00010000,
                 (int)(32u  << 16),     // tensor_dim0[15:0]  @ bits 63:48
                 (int)(128u << 16),     // tensor_dim1[15:0]  @ bits 95:80
                 (int)(32u  << 16),     // tile_dim0          @ bits 127:112
                 128,                   // tile_dim1          @ bits 143:128
                 strideElems,           // tensor_dim0_stride @ bits 191:160
                 0, 0 };
  int32x4 gz = {};                      // groups 2/3 unused (2D tile)
#if defined(__clang_major__) && (__clang_major__ >= 23)
  int32x8 gz8 = {};
  __builtin_amdgcn_tensor_load_to_lds(g0, g1, gz, gz, gz8, 0);
#else
  __builtin_amdgcn_tensor_load_to_lds(g0, g1, gz, gz, 0);
#endif
}
__device__ __forceinline__ unsigned lds_off(const void* p) {
  return (unsigned)(uintptr_t)p;        // low 32 bits of generic addr = LDS offset
}

// ---------------------------------------------------------------------------
// LayerNorm over H=1024, f32 in -> bf16 out.  One block per row, 256 threads.
// ---------------------------------------------------------------------------
__global__ __launch_bounds__(256)
void k_ln(const float* __restrict__ x, const float* __restrict__ g,
          const float* __restrict__ b, __bf16* __restrict__ out) {
  const int row = blockIdx.x;
  const int tid = threadIdx.x;
  const float4 v = ((const float4*)(x + (size_t)row * HH))[tid];
  float s  = v.x + v.y + v.z + v.w;
  float s2 = v.x*v.x + v.y*v.y + v.z*v.z + v.w*v.w;
  #pragma unroll
  for (int m = 16; m >= 1; m >>= 1) {       // wave32 reduce
    s  += __shfl_xor(s,  m);
    s2 += __shfl_xor(s2, m);
  }
  __shared__ float red[16];
  const int wav = tid >> 5, lane = tid & 31;
  if (lane == 0) { red[wav] = s; red[8 + wav] = s2; }
  __syncthreads();
  float ts = 0.f, ts2 = 0.f;
  #pragma unroll
  for (int i = 0; i < 8; ++i) { ts += red[i]; ts2 += red[8 + i]; }
  const float mu   = ts * (1.0f / HH);
  const float var  = ts2 * (1.0f / HH) - mu * mu;
  const float rstd = rsqrtf(var + 1e-6f);
  const float4 gg = ((const float4*)g)[tid];
  const float4 bb = ((const float4*)b)[tid];
  bf16x4 o;
  o[0] = (__bf16)((v.x - mu) * rstd * gg.x + bb.x);
  o[1] = (__bf16)((v.y - mu) * rstd * gg.y + bb.y);
  o[2] = (__bf16)((v.z - mu) * rstd * gg.z + bb.z);
  o[3] = (__bf16)((v.w - mu) * rstd * gg.w + bb.w);
  ((bf16x4*)(out + (size_t)row * HH))[tid] = o;
}

// ---------------------------------------------------------------------------
// Transpose+convert: W[K][N] f32 -> Wt[N][K] bf16, 32x32 LDS tiles.
// ---------------------------------------------------------------------------
__global__ __launch_bounds__(256)
void k_cvtT(const float* __restrict__ W, __bf16* __restrict__ Wt, int K, int N) {
  __shared__ float t[32][33];
  const int n0 = blockIdx.x * 32, k0 = blockIdx.y * 32;
  const int tx = threadIdx.x, ty = threadIdx.y;
  for (int i = ty; i < 32; i += 8)
    t[i][tx] = W[(size_t)(k0 + i) * N + n0 + tx];
  __syncthreads();
  for (int i = ty; i < 32; i += 8)
    Wt[(size_t)(n0 + i) * K + k0 + tx] = (__bf16)t[tx][i];
}

// ---------------------------------------------------------------------------
// WMMA GEMM: C[M,N] = A[M,K]bf16 @ Bt[N,K]bf16^T (+bias, epilogue variants).
// 256 thr = 8 waves; macro-tile 128x128; wave tile 64x32 (4x2 WMMA frags).
// Tiles are DMA'd by the TDM: wave 0 drives the A tile, wave 1 the B tile,
// double-buffered; compute on buffer i overlaps the DMA of buffer i+1.
// ---------------------------------------------------------------------------
enum { EPI_QKV = 0, EPI_VT = 1, EPI_GELU = 2, EPI_RES = 3 };

template<int EPI>
__global__ __launch_bounds__(256)
void k_gemm(const __bf16* __restrict__ A, const __bf16* __restrict__ Bt,
            const float* __restrict__ bias, const float* __restrict__ resid,
            void* __restrict__ outp, int M, int N, int K) {
  __shared__ alignas(16) __bf16 As[2][128 * 32];
  __shared__ alignas(16) __bf16 Bs[2][128 * 32];
  const int tid = threadIdx.x, lane = tid & 31, wav = tid >> 5;
  const int wm = wav >> 2, wn = wav & 3;
  const int tileM = blockIdx.y * 128, tileN = blockIdx.x * 128;
  const int l15 = lane & 15, hi16 = (lane & 16) >> 4;
  const int kh = hi16 * 8, K0 = hi16 * 16;      // ISA A-/B-fragment K offsets

  f32x8 acc[4][2] = {};

  const __bf16* Agt = A  + (size_t)tileM * K;   // tile base rows, stride K
  const __bf16* Bgt = Bt + (size_t)tileN * K;
  const int nk = K / 32;

  // Prologue: DMA buffer 0.
  if (wav == 0) tdm_load_tile_128x32(Agt, lds_off(&As[0][0]), K);
  if (wav == 1) tdm_load_tile_128x32(Bgt, lds_off(&Bs[0][0]), K);

  for (int i = 0; i < nk; ++i) {
    const int cur = i & 1;
    if (i + 1 < nk) {                           // kick DMA for next buffer
      if (wav == 0) tdm_load_tile_128x32(Agt + (i + 1) * 32,
                                         lds_off(&As[cur ^ 1][0]), K);
      if (wav == 1) tdm_load_tile_128x32(Bgt + (i + 1) * 32,
                                         lds_off(&Bs[cur ^ 1][0]), K);
      if (wav < 2) __builtin_amdgcn_s_wait_tensorcnt(1);  // cur done (in-order)
    } else {
      if (wav < 2) __builtin_amdgcn_s_wait_tensorcnt(0);
    }
    __syncthreads();                            // publish cur buffer

    const __bf16* Ac = As[cur];
    const __bf16* Bc = Bs[cur];
    bf16x16 fb[2];
    #pragma unroll
    for (int nt = 0; nt < 2; ++nt) {
      const int nrow = wn * 32 + nt * 16 + l15;
      bf16x8 lo = *(const bf16x8*)&Bc[nrow * 32 + kh];
      bf16x8 hi = *(const bf16x8*)&Bc[nrow * 32 + kh + 16];
      fb[nt] = cat8(lo, hi);
    }
    #pragma unroll
    for (int mt = 0; mt < 4; ++mt) {
      const int arow = wm * 64 + mt * 16 + l15;
      bf16x8 lo = *(const bf16x8*)&Ac[arow * 32 + kh];
      bf16x8 hi = *(const bf16x8*)&Ac[arow * 32 + kh + 16];
      bf16x16 fa = cat8(lo, hi);
      #pragma unroll
      for (int nt = 0; nt < 2; ++nt)
        acc[mt][nt] = wmma_bf16(fa, fb[nt], acc[mt][nt]);
    }
    __syncthreads();                            // reads done before DMA reuses cur
  }

  // Epilogue: C layout -> lane holds rows (hi16*8 + r), col l15 per frag.
  const int rowoff = hi16 * 8;
  #pragma unroll
  for (int mt = 0; mt < 4; ++mt) {
    #pragma unroll
    for (int nt = 0; nt < 2; ++nt) {
      const int gn = tileN + wn * 32 + nt * 16 + l15;
      const float bv = bias[gn];
      #pragma unroll
      for (int r = 0; r < 8; ++r) {
        const int gm = tileM + wm * 64 + mt * 16 + rowoff + r;
        float v = acc[mt][nt][r] + bv;
        if constexpr (EPI == EPI_RES) {
          float* out = (float*)outp;
          out[(size_t)gm * N + gn] = v + resid[(size_t)gm * N + gn];
        } else if constexpr (EPI == EPI_GELU) {
          __bf16* out = (__bf16*)outp;
          const float gel = 0.5f * v * (1.0f + erff(v * 0.70710678118654752f));
          out[(size_t)gm * N + gn] = (__bf16)gel;
        } else if constexpr (EPI == EPI_QKV) {   // [B,NH,S,HD]
          __bf16* out = (__bf16*)outp;
          const int b = gm >> 10, s = gm & (SS - 1);
          const int h = gn >> 6,  d = gn & (HDD - 1);
          out[(((size_t)(b * NHH + h)) * SS + s) * HDD + d] = (__bf16)v;
        } else {                                 // EPI_VT: [B,NH,HD,S]
          __bf16* out = (__bf16*)outp;
          const int b = gm >> 10, s = gm & (SS - 1);
          const int h = gn >> 6,  d = gn & (HDD - 1);
          out[(((size_t)(b * NHH + h)) * HDD + d) * SS + s] = (__bf16)v;
        }
      }
    }
  }
}

// ---------------------------------------------------------------------------
// Flash attention: one (b,h) + 64 query rows per block (4 waves x 16 rows).
// scores = Q K^T via WMMA, online softmax on accumulators, P V^T via WMMA.
// ---------------------------------------------------------------------------
__global__ __launch_bounds__(128)
void k_attn(const __bf16* __restrict__ Q, const __bf16* __restrict__ Kb,
            const __bf16* __restrict__ VT, __bf16* __restrict__ Ob) {
  __shared__ alignas(16) __bf16 Pl[4][16 * 32];   // per-wave P bounce buffer
  const int bh  = blockIdx.x;                     // b*NH + h
  const int qt  = blockIdx.y;
  const int tid = threadIdx.x, lane = tid & 31, wav = tid >> 5;
  const int l15 = lane & 15, hi16 = (lane & 16) >> 4;
  const int kh = hi16 * 8, K0 = hi16 * 16;
  const int qbase = qt * 64 + wav * 16;
  const size_t base = (size_t)bh * SS * HDD;

  // Q fragments (persist over the key loop): A-layout, row = l15.
  bf16x16 fq[2];
  {
    const __bf16* qrow = Q + base + (size_t)(qbase + l15) * HDD;
    #pragma unroll
    for (int ks = 0; ks < 2; ++ks) {
      bf16x8 lo = *(const bf16x8*)(qrow + ks * 32 + kh);
      bf16x8 hi = *(const bf16x8*)(qrow + ks * 32 + kh + 16);
      fq[ks] = cat8(lo, hi);
    }
  }

  f32x8 accO[4] = {};
  float mrow[8], lrow[8];
  #pragma unroll
  for (int r = 0; r < 8; ++r) { mrow[r] = -1e30f; lrow[r] = 0.f; }

  for (int kt = 0; kt < SS / 32; ++kt) {
    const int keyb = kt * 32;
    // scores[16 q x 32 keys] = Q @ K^T  (B^T == K, rows load contiguously)
    f32x8 sA[2] = {};
    #pragma unroll
    for (int nt = 0; nt < 2; ++nt) {
      const __bf16* krow = Kb + base + (size_t)(keyb + nt * 16 + l15) * HDD;
      #pragma unroll
      for (int ks = 0; ks < 2; ++ks) {
        bf16x8 lo = *(const bf16x8*)(krow + ks * 32 + kh);
        bf16x8 hi = *(const bf16x8*)(krow + ks * 32 + kh + 16);
        sA[nt] = wmma_bf16(fq[ks], cat8(lo, hi), sA[nt]);
      }
    }
    // Online softmax (rows replicated across 16-lane halves).
    #pragma unroll
    for (int r = 0; r < 8; ++r) {
      const float v0 = sA[0][r] * 0.125f;        // 1/sqrt(HD)
      const float v1 = sA[1][r] * 0.125f;
      float mx = fmaxf(v0, v1);
      #pragma unroll
      for (int m = 8; m >= 1; m >>= 1) mx = fmaxf(mx, __shfl_xor(mx, m));
      const float mnew = fmaxf(mrow[r], mx);
      const float p0 = __expf(v0 - mnew);
      const float p1 = __expf(v1 - mnew);
      float ps = p0 + p1;
      #pragma unroll
      for (int m = 8; m >= 1; m >>= 1) ps += __shfl_xor(ps, m);
      const float corr = __expf(mrow[r] - mnew);
      lrow[r] = lrow[r] * corr + ps;
      mrow[r] = mnew;
      #pragma unroll
      for (int o = 0; o < 4; ++o) accO[o][r] *= corr;
      const int prow = hi16 * 8 + r;             // C-layout -> LDS (A-layout src)
      Pl[wav][prow * 32 + l15]      = (__bf16)p0;
      Pl[wav][prow * 32 + 16 + l15] = (__bf16)p1;
    }
    // P (A-layout from LDS; DS ops are in-order per wave) @ V^T
    {
      bf16x8 lo = *(const bf16x8*)&Pl[wav][l15 * 32 + kh];
      bf16x8 hi = *(const bf16x8*)&Pl[wav][l15 * 32 + kh + 16];
      const bf16x16 fp = cat8(lo, hi);
      #pragma unroll
      for (int o = 0; o < 4; ++o) {
        const __bf16* vrow = VT + base + (size_t)(o * 16 + l15) * SS + keyb + kh;
        bf16x8 vlo = *(const bf16x8*)vrow;
        bf16x8 vhi = *(const bf16x8*)(vrow + 16);
        accO[o] = wmma_bf16(fp, cat8(vlo, vhi), accO[o]);
      }
    }
  }

  // O /= l, store back to [B,S,H] (h*64+d) as bf16.
  const int b = bh >> 4, h = bh & 15;
  #pragma unroll
  for (int o = 0; o < 4; ++o) {
    const int d = h * HDD + o * 16 + l15;
    #pragma unroll
    for (int r = 0; r < 8; ++r) {
      const int srow = qbase + hi16 * 8 + r;
      const float ov = accO[o][r] / lrow[r];
      Ob[((size_t)(b * SS + srow)) * HH + d] = (__bf16)ov;
    }
  }
}

// ---------------------------------------------------------------------------
// Host side.  Workspace layout (136 MB):
//   [0,8M)    Wq/Wk/Wv/Wo^T bf16    [8M,16M)  W1^T bf16   [16M,24M) W2^T bf16
//   [24M,40M) h / h2 bf16
//   [40M,56M) Q | [56M,72M) K | [72M,88M) V^T | [88M,104M) attn  (bf16)
//   [104M,136M) x2 f32
//   MLP activation (64 MB bf16) reuses [40M,104M) after attention is consumed.
// ---------------------------------------------------------------------------
extern "C" void kernel_launch(void* const* d_in, const int* in_sizes, int n_in,
                              void* d_out, int out_size, void* d_ws, size_t ws_size,
                              hipStream_t stream) {
  const float* x    = (const float*)d_in[0];
  const float* ln1g = (const float*)d_in[1];
  const float* ln1b = (const float*)d_in[2];
  const float* Wq   = (const float*)d_in[3];
  const float* bq   = (const float*)d_in[4];
  const float* Wk   = (const float*)d_in[5];
  const float* bk   = (const float*)d_in[6];
  const float* Wv   = (const float*)d_in[7];
  const float* bv   = (const float*)d_in[8];
  const float* Wo   = (const float*)d_in[9];
  const float* bo   = (const float*)d_in[10];
  const float* ln2g = (const float*)d_in[11];
  const float* ln2b = (const float*)d_in[12];
  const float* W1   = (const float*)d_in[13];
  const float* b1   = (const float*)d_in[14];
  const float* W2   = (const float*)d_in[15];
  const float* b2   = (const float*)d_in[16];
  float* out = (float*)d_out;

  char* ws = (char*)d_ws;
  const size_t MB = 1ull << 20;
  __bf16* Wqt  = (__bf16*)(ws);
  __bf16* Wkt  = (__bf16*)(ws + 2 * MB);
  __bf16* Wvt  = (__bf16*)(ws + 4 * MB);
  __bf16* Wot  = (__bf16*)(ws + 6 * MB);
  __bf16* W1t  = (__bf16*)(ws + 8 * MB);
  __bf16* W2t  = (__bf16*)(ws + 16 * MB);
  __bf16* Hb   = (__bf16*)(ws + 24 * MB);
  __bf16* Qb   = (__bf16*)(ws + 40 * MB);
  __bf16* Kbf  = (__bf16*)(ws + 56 * MB);
  __bf16* VTb  = (__bf16*)(ws + 72 * MB);
  __bf16* Attn = (__bf16*)(ws + 88 * MB);
  float*  X2   = (float*) (ws + 104 * MB);
  __bf16* Mlp  = (__bf16*)(ws + 40 * MB);      // reuse of Q/K/VT/attn region

  const dim3 cb(32, 8);
  k_cvtT<<<dim3(HH / 32,   HH / 32),  cb, 0, stream>>>(Wq, Wqt, HH,   HH);
  k_cvtT<<<dim3(HH / 32,   HH / 32),  cb, 0, stream>>>(Wk, Wkt, HH,   HH);
  k_cvtT<<<dim3(HH / 32,   HH / 32),  cb, 0, stream>>>(Wv, Wvt, HH,   HH);
  k_cvtT<<<dim3(HH / 32,   HH / 32),  cb, 0, stream>>>(Wo, Wot, HH,   HH);
  k_cvtT<<<dim3(MLPD / 32, HH / 32),  cb, 0, stream>>>(W1, W1t, HH,   MLPD);
  k_cvtT<<<dim3(HH / 32,   MLPD / 32),cb, 0, stream>>>(W2, W2t, MLPD, HH);

  k_ln<<<MM, 256, 0, stream>>>(x, ln1g, ln1b, Hb);

  const dim3 gH(HH / 128, MM / 128);           // (8, 64)
  k_gemm<EPI_QKV><<<gH, 256, 0, stream>>>(Hb, Wqt, bq, nullptr, Qb,  MM, HH, HH);
  k_gemm<EPI_QKV><<<gH, 256, 0, stream>>>(Hb, Wkt, bk, nullptr, Kbf, MM, HH, HH);
  k_gemm<EPI_VT ><<<gH, 256, 0, stream>>>(Hb, Wvt, bv, nullptr, VTb, MM, HH, HH);

  k_attn<<<dim3(BB * NHH, SS / 64), 128, 0, stream>>>(Qb, Kbf, VTb, Attn);

  k_gemm<EPI_RES><<<gH, 256, 0, stream>>>(Attn, Wot, bo, x, X2, MM, HH, HH);

  k_ln<<<MM, 256, 0, stream>>>(X2, ln2g, ln2b, Hb);

  k_gemm<EPI_GELU><<<dim3(MLPD / 128, MM / 128), 256, 0, stream>>>(
      Hb, W1t, b1, nullptr, Mlp, MM, MLPD, HH);

  k_gemm<EPI_RES><<<gH, 256, 0, stream>>>(Mlp, W2t, b2, X2, out, MM, HH, MLPD);

  (void)in_sizes; (void)n_in; (void)out_size; (void)ws_size;
}
